// ResGatedGCNConv_13073880449502
// MI455X (gfx1250) — matrix operand order repaired and checked by reading, not compiled, not run on hardware
//
#include <hip/hip_runtime.h>
#include <hip/hip_bf16.h>

typedef __attribute__((ext_vector_type(16))) __bf16 v16bf;
typedef __attribute__((ext_vector_type(8)))  __bf16 v8bf;
typedef __attribute__((ext_vector_type(2)))  __bf16 v2bf;
typedef __attribute__((ext_vector_type(8)))  float  v8f;

#define N_NODES 40000
#define N_EDGES 640000
#define DIM     128

// ---------------------------------------------------------------------------
// Kernel 0: pack the 4 weight matrices into bf16, column-major:
//   Wp[m][c][k] = (bf16) W_m[k*128 + c]     (4 * 128 * 128 * 2 B = 128 KB)
// After packing, a lane's B fragment (16 contiguous K values of one column,
// pairwise-packed) is exactly 32 contiguous bytes -> 2x global_load_b128.
// ---------------------------------------------------------------------------
__global__ __launch_bounds__(256) void pack_weights_kernel(
    const float* __restrict__ Wk, const float* __restrict__ Wq,
    const float* __restrict__ Wv, const float* __restrict__ Ws,
    __bf16* __restrict__ Wp)
{
    const int t = blockIdx.x * 256 + threadIdx.x;   // 0 .. 65535
    const int m   = t >> 14;
    const int rem = t & 16383;
    const int c   = rem >> 7;
    const int k   = rem & 127;
    const float* W = (m == 0) ? Wk : (m == 1) ? Wq : (m == 2) ? Wv : Ws;
    Wp[t] = (__bf16)W[k * DIM + c];
}

// ---------------------------------------------------------------------------
// Kernel 1: fused 4-way GEMM.  One block = one 16-row tile of x, 8 waves.
//   x tile converted to bf16 in LDS during the coalesced stage; A fragments
//   are 2x ds_load_b128, B fragments 2x global_load_b128 from packed weights.
//   16 static v_wmma_f32_16x16x32_bf16 per wave (4 matrices x 4 K-chunks).
// ---------------------------------------------------------------------------
__global__ __launch_bounds__(256) void gemm_kqvs_kernel(
    const float* __restrict__ x,
    const __bf16* __restrict__ Wp,
    const float* __restrict__ bk, const float* __restrict__ bq,
    const float* __restrict__ bv, const float* __restrict__ bs,
    const float* __restrict__ broot,
    float* __restrict__ kbuf, float* __restrict__ qbuf,
    float* __restrict__ vbuf, float* __restrict__ out)
{
    __shared__ __bf16 xs[16 * DIM];   // 4 KB, bf16 row-major tile

    const int tid     = threadIdx.x;
    const int rowBase = blockIdx.x * 16;             // N divisible by 16

    // Coalesced stage: each thread loads 4 float2 (8 floats), converts to
    // packed bf16 pairs, stores to LDS.
#pragma unroll
    for (int r = 0; r < 4; ++r) {
        const int pidx = tid + 256 * r;              // pair index 0..1023
        const float2 f = *(const float2*)(x + (size_t)rowBase * DIM + 2 * pidx);
        v2bf p;
        p[0] = (__bf16)f.x;
        p[1] = (__bf16)f.y;
        *(v2bf*)(&xs[2 * pidx]) = p;
    }
    __syncthreads();

    const int wave = tid >> 5;
    const int lane = tid & 31;
    const int half = lane >> 4;       // 0: lanes 0-15, 1: lanes 16-31
    const int l16  = lane & 15;
    const int col  = wave * 16 + l16; // this wave's output column slab

    // Per-lane packed-weight column bases (128 bf16 per column, contiguous K).
    const v8bf* bpK = (const v8bf*)(Wp + (size_t)(0 * 16384 + col * DIM));
    const v8bf* bpQ = (const v8bf*)(Wp + (size_t)(1 * 16384 + col * DIM));
    const v8bf* bpV = (const v8bf*)(Wp + (size_t)(2 * 16384 + col * DIM));
    const v8bf* bpS = (const v8bf*)(Wp + (size_t)(3 * 16384 + col * DIM));
    const v8bf* arow = (const v8bf*)(&xs[l16 * DIM]);

    v8f accK = {}; v8f accQ = {}; v8f accV = {}; v8f accS = {};

#pragma unroll
    for (int kc = 0; kc < 4; ++kc) {
        const int Kb8 = kc * 4;       // Kb / 8

        // A fragment: two contiguous 8-value K runs -> 2x ds_load_b128
        const v8bf a0 = arow[Kb8 + 2 * half];
        const v8bf a1 = arow[Kb8 + 2 * half + 1];
        const v16bf a = __builtin_shufflevector(a0, a1,
            0, 1, 2, 3, 4, 5, 6, 7, 8, 9, 10, 11, 12, 13, 14, 15);

        // B fragments: 16 contiguous K values per lane -> 2x global_load_b128
        const int bi = Kb8 + 2 * half;
        const v8bf k0 = bpK[bi], k1 = bpK[bi + 1];
        const v8bf q0 = bpQ[bi], q1 = bpQ[bi + 1];
        const v8bf v0 = bpV[bi], v1 = bpV[bi + 1];
        const v8bf s0 = bpS[bi], s1 = bpS[bi + 1];
        const v16bf fbK = __builtin_shufflevector(k0, k1,
            0, 1, 2, 3, 4, 5, 6, 7, 8, 9, 10, 11, 12, 13, 14, 15);
        const v16bf fbQ = __builtin_shufflevector(q0, q1,
            0, 1, 2, 3, 4, 5, 6, 7, 8, 9, 10, 11, 12, 13, 14, 15);
        const v16bf fbV = __builtin_shufflevector(v0, v1,
            0, 1, 2, 3, 4, 5, 6, 7, 8, 9, 10, 11, 12, 13, 14, 15);
        const v16bf fbS = __builtin_shufflevector(s0, s1,
            0, 1, 2, 3, 4, 5, 6, 7, 8, 9, 10, 11, 12, 13, 14, 15);

        accK = __builtin_amdgcn_wmma_f32_16x16x32_bf16(false, a, false, fbK,
                                                       (short)0, accK, false, false);
        accQ = __builtin_amdgcn_wmma_f32_16x16x32_bf16(false, a, false, fbQ,
                                                       (short)0, accQ, false, false);
        accV = __builtin_amdgcn_wmma_f32_16x16x32_bf16(false, a, false, fbV,
                                                       (short)0, accV, false, false);
        accS = __builtin_amdgcn_wmma_f32_16x16x32_bf16(false, a, false, fbS,
                                                       (short)0, accS, false, false);
    }

    // C/D layout: VGPR r holds (M = r + 8*half, N = l16).
    const float bkc = bk[col];
    const float bqc = bq[col];
    const float bvc = bv[col];
    const float bsc = bs[col] + broot[col];
#pragma unroll
    for (int r = 0; r < 8; ++r) {
        const int row    = rowBase + r + 8 * half;
        const size_t off = (size_t)row * DIM + col;
        kbuf[off] = accK[r] + bkc;
        qbuf[off] = accQ[r] + bqc;
        vbuf[off] = accV[r] + bvc;
        out[off]  = accS[r] + bsc;
    }
}

// ---------------------------------------------------------------------------
// Kernel 2: edge messages.  32 lanes per edge, float4 per lane (128 feats).
//   gate = sigmoid(k[dst] + q[src]);  out[dst] += gate * v[src]
// k/q/v buffers (61 MB) live in the 192 MB L2; atomics resolve in L2.
// ---------------------------------------------------------------------------
__global__ __launch_bounds__(256) void edge_msg_kernel(
    const int* __restrict__ ei,
    const float* __restrict__ kbuf, const float* __restrict__ qbuf,
    const float* __restrict__ vbuf, float* __restrict__ out)
{
    const long gtid = (long)blockIdx.x * blockDim.x + threadIdx.x;
    const int  e    = (int)(gtid >> 5);
    if (e >= N_EDGES) return;
    const int lane = threadIdx.x & 31;
    const int f    = lane * 4;

    const int src = ei[e];             // edge_index[0, e]
    const int dst = ei[N_EDGES + e];   // edge_index[1, e]

    const float4 kv = *(const float4*)(kbuf + (size_t)dst * DIM + f);
    const float4 qv = *(const float4*)(qbuf + (size_t)src * DIM + f);
    const float4 vv = *(const float4*)(vbuf + (size_t)src * DIM + f);

    float4 m;
    m.x = vv.x / (1.0f + __expf(-(kv.x + qv.x)));
    m.y = vv.y / (1.0f + __expf(-(kv.y + qv.y)));
    m.z = vv.z / (1.0f + __expf(-(kv.z + qv.z)));
    m.w = vv.w / (1.0f + __expf(-(kv.w + qv.w)));

    float* o = out + (size_t)dst * DIM + f;
    unsafeAtomicAdd(o + 0, m.x);   // -> global_atomic_add_f32
    unsafeAtomicAdd(o + 1, m.y);
    unsafeAtomicAdd(o + 2, m.z);
    unsafeAtomicAdd(o + 3, m.w);
}

// ---------------------------------------------------------------------------
extern "C" void kernel_launch(void* const* d_in, const int* in_sizes, int n_in,
                              void* d_out, int out_size, void* d_ws, size_t ws_size,
                              hipStream_t stream)
{
    const float* x   = (const float*)d_in[0];
    const int*   ei  = (const int*)  d_in[1];   // [2, E] int32
    const float* Wk  = (const float*)d_in[2];
    const float* bk  = (const float*)d_in[3];
    const float* Wq  = (const float*)d_in[4];
    const float* bq  = (const float*)d_in[5];
    const float* Wv  = (const float*)d_in[6];
    const float* bv  = (const float*)d_in[7];
    const float* Ws  = (const float*)d_in[8];
    const float* bs  = (const float*)d_in[9];
    const float* b   = (const float*)d_in[10];
    float* out = (float*)d_out;

    // Workspace layout:
    //   [0, 128KB)          packed bf16 weights (4 x 128 x 128)
    //   [128KB, +61.44MB)   k, q, v f32 buffers
    __bf16* Wp  = (__bf16*)d_ws;
    float* kbuf = (float*)((char*)d_ws + 4 * DIM * DIM * sizeof(__bf16));
    float* qbuf = kbuf + (size_t)N_NODES * DIM;
    float* vbuf = qbuf + (size_t)N_NODES * DIM;

    // Phase 0: pack weights to bf16 column-major (65536 elems, 256 blocks).
    pack_weights_kernel<<<256, 256, 0, stream>>>(Wk, Wq, Wv, Ws, Wp);

    // Phase 1: fused GEMMs (2500 row-tiles of 16, 8 waves each).
    gemm_kqvs_kernel<<<N_NODES / 16, 256, 0, stream>>>(
        x, Wp, bk, bq, bv, bs, b, kbuf, qbuf, vbuf, out);

    // Phase 2: edge messages + scatter-add.  32 lanes/edge.
    const int threads = 256;
    const long total  = (long)N_EDGES * 32;
    const int  blocks = (int)((total + threads - 1) / threads);   // 80000
    edge_msg_kernel<<<blocks, threads, 0, stream>>>(ei, kbuf, qbuf, vbuf, out);
}
